// Qwen3Attention_62491774157479
// MI455X (gfx1250) — compile-verified, matrix operand order checked by reading, not run on hardware
//
#include <hip/hip_runtime.h>
#include <hip/hip_bf16.h>

typedef __attribute__((ext_vector_type(16))) __bf16 v16bf;
typedef __attribute__((ext_vector_type(8)))  float  v8f;

namespace {
constexpr int B = 2, S = 2048, HID = 2560;
constexpr int NH = 32, NKV = 8, HD = 128;
constexpr int GROUPS = NH / NKV;      // 4
constexpr int M = B * S;              // 4096 rows
constexpr float EPS = 1e-6f;
constexpr float SCALE = 0.08838834764831845f; // 1/sqrt(128)

constexpr int AP = 40;   // LDS row stride (bf16) for 32-deep K tiles (bank pad)
constexpr int FP = 136;  // LDS row stride (bf16) for 128-deep tiles
constexpr int CP = 132;  // LDS row stride (f32) for 128-wide C tiles
constexpr int TILE = 128 * AP;  // bf16 elems per 128x32 LDS tile
}

// 16-element bf16 A/B fragment for v_wmma_f32_16x16x32_bf16.
// Lanes 0-15 hold K {kb..kb+7} and {kb+16..kb+23}; lanes 16-31 use kb+8.
__device__ __forceinline__ v16bf ld_frag(const __bf16* p, int kb) {
  v16bf f;
#pragma unroll
  for (int i = 0; i < 8; ++i) { f[i] = p[kb + i]; f[i + 8] = p[kb + 16 + i]; }
  return f;
}

__device__ __forceinline__ v8f wmma_bf16(v16bf a, v16bf b, v8f c) {
  return __builtin_amdgcn_wmma_f32_16x16x32_bf16(false, a, false, b, (short)0, c,
                                                 false, false);
}

// CDNA5 async global->LDS copy: 16 bytes per lane, tracked by ASYNCcnt.
__device__ __forceinline__ void async_copy16(const __bf16* gsrc, __bf16* ldst) {
  unsigned lds = (unsigned)(uintptr_t)ldst;
  unsigned long long g = (unsigned long long)(uintptr_t)gsrc;
  asm volatile("global_load_async_to_lds_b128 %0, %1, off"
               :: "v"(lds), "v"(g) : "memory");
}
__device__ __forceinline__ void wait_async0() {
  asm volatile("s_wait_asynccnt 0x0" ::: "memory");
}

// ---------------------------------------------------------------------------
// Kernel 1: QKV projection GEMM, 128 rows x one full head, double-buffered LDS.
// mode 0/1: fused RMSNorm + RoPE epilogue (q/k), mode 2: plain bf16 store (v).
// Output layout: [B, nheads, S, HD] bf16.
// ---------------------------------------------------------------------------
__global__ __launch_bounds__(256) void qkv_gemm_kernel(
    const float* __restrict__ Ag, const float* __restrict__ Wg,
    const float* __restrict__ normw, const int* __restrict__ posids,
    __bf16* __restrict__ outg, int nheads, int mode) {
  extern __shared__ char smem[];
  __bf16* As0 = (__bf16*)smem;               // [2][128][AP]
  __bf16* Bs0 = As0 + 2 * TILE;              // [2][128 cols][AP] (Bs[col][k])
  float*  Cs  = (float*)(Bs0 + 2 * TILE);    // [128][CP], modes 0/1 only

  const int tid = threadIdx.x;
  const int w = tid >> 5, lane = tid & 31;
  const int hf = lane >> 4, r16 = lane & 15;
  const int m0 = blockIdx.x * 128;
  const int h  = blockIdx.y;
  const int Wld = nheads * HD;

  v8f acc[8] = {};

  auto load_tiles = [&](int k0, int buf) {
    __bf16* As = As0 + buf * TILE;
    __bf16* Bs = Bs0 + buf * TILE;
#pragma unroll
    for (int it = 0; it < 4; ++it) {
      int slot = it * 256 + tid;
      int r = slot >> 3, c4 = (slot & 7) * 4;
      float4 v = *(const float4*)(Ag + (size_t)(m0 + r) * HID + k0 + c4);
      __bf16* dst = As + r * AP + c4;
      dst[0] = (__bf16)v.x; dst[1] = (__bf16)v.y;
      dst[2] = (__bf16)v.z; dst[3] = (__bf16)v.w;
    }
#pragma unroll
    for (int it = 0; it < 4; ++it) {
      int slot = it * 256 + tid;
      int kk = slot >> 5, c4 = (slot & 31) * 4;
      float4 v = *(const float4*)(Wg + (size_t)(k0 + kk) * Wld + h * HD + c4);
      Bs[(c4 + 0) * AP + kk] = (__bf16)v.x;
      Bs[(c4 + 1) * AP + kk] = (__bf16)v.y;
      Bs[(c4 + 2) * AP + kk] = (__bf16)v.z;
      Bs[(c4 + 3) * AP + kk] = (__bf16)v.w;
    }
    if (k0 + 32 < HID)  // gfx1250 global_prefetch_b8 for the tile after next
      __builtin_prefetch(Wg + (size_t)(k0 + 32) * Wld + h * HD, 0, 1);
  };

  auto compute = [&](int buf) {
    const __bf16* As = As0 + buf * TILE;
    const __bf16* Bs = Bs0 + buf * TILE;
    const int kb = hf * 8;
    v16bf af = ld_frag(As + (16 * w + r16) * AP, kb);
    v16bf bfr[8];
#pragma unroll
    for (int nt = 0; nt < 8; ++nt)
      bfr[nt] = ld_frag(Bs + (16 * nt + r16) * AP, kb);
#pragma unroll
    for (int nt = 0; nt < 8; ++nt)   // back-to-back WMMA issue
      acc[nt] = wmma_bf16(af, bfr[nt], acc[nt]);
  };

  load_tiles(0, 0);
  int buf = 0;
  for (int k0 = 0; k0 < HID; k0 += 32) {
    __syncthreads();                       // tile[buf] ready; buf^1 free
    if (k0 + 32 < HID) load_tiles(k0 + 32, buf ^ 1);
    compute(buf);
    buf ^= 1;
  }

  if (mode == 2) {  // V: direct bf16 store
#pragma unroll
    for (int nt = 0; nt < 8; ++nt)
#pragma unroll
      for (int r = 0; r < 8; ++r) {
        int row = m0 + 16 * w + r + 8 * hf;
        int bb = row / S, ss = row % S;
        outg[(((size_t)bb * nheads + h) * S + ss) * HD + 16 * nt + r16] =
            (__bf16)acc[nt][r];
      }
    return;
  }

  // Dump C (f32) to LDS so each row's full head dim is visible for RMSNorm/RoPE.
  __syncthreads();
#pragma unroll
  for (int nt = 0; nt < 8; ++nt)
#pragma unroll
    for (int r = 0; r < 8; ++r)
      Cs[(16 * w + r + 8 * hf) * CP + 16 * nt + r16] = acc[nt][r];
  __syncthreads();

  if (tid < 128) {
    int row = m0 + tid;
    int bb = row / S, ss = row % S;
    float* crow = Cs + tid * CP;
    float ssq = 0.0f;
    for (int c = 0; c < HD; ++c) { float x = crow[c]; ssq += x * x; }
    float inv = rsqrtf(ssq * (1.0f / HD) + EPS);
    for (int c = 0; c < HD; ++c) crow[c] = crow[c] * inv * normw[c];
    float pos = (float)posids[bb * S + ss];
    __bf16* orow = outg + (((size_t)bb * nheads + h) * S + ss) * HD;
    for (int c = 0; c < HD; ++c) {
      float invf = __powf(1.0e6f, -(float)(c & 63) * (1.0f / 64.0f));
      float sn, cs;
      __sincosf(pos * invf, &sn, &cs);
      float xn = crow[c];
      float xp = crow[c ^ 64];
      float rot = (c < 64) ? -xp : xp;
      orow[c] = (__bf16)(xn * cs + rot * sn);
    }
  }
}

// ---------------------------------------------------------------------------
// Kernel 2: causal flash attention per (b, head, 128-query tile), GQA.
// Q/K tiles staged via global_load_async_to_lds_b128 (ASYNCcnt);
// scores and P.V via v_wmma_f32_16x16x32_bf16; online softmax in f32 LDS.
// ---------------------------------------------------------------------------
__global__ __launch_bounds__(256) void attn_kernel(
    const __bf16* __restrict__ Q, const __bf16* __restrict__ Kg,
    const __bf16* __restrict__ Vg, __bf16* __restrict__ Og) {
  extern __shared__ char smem[];
  __bf16* Qs = (__bf16*)smem;            // [128 q][FP] hd-contiguous
  __bf16* Ks = Qs + 128 * FP;            // [128 key][FP] hd-contiguous
  __bf16* Vt = Ks + 128 * FP;            // [128 hd][FP] key-contiguous
  float*  Sc = (float*)(Vt + 128 * FP);  // [128 q][CP] scores f32
  __bf16* Ps = (__bf16*)(Sc + 128 * CP); // [128 q][FP] probs bf16
  float*  mrow = (float*)(Ps + 128 * FP);
  float*  lrow = mrow + 128;
  float*  srow = lrow + 128;

  const int tid = threadIdx.x;
  const int w = tid >> 5, lane = tid & 31;
  const int hf = lane >> 4, r16 = lane & 15;
  const int qt = blockIdx.x, h = blockIdx.y, b = blockIdx.z;
  const int kvh = h / GROUPS;
  const int q0 = qt * 128;
  const __bf16* qbase = Q + (((size_t)b * NH + h) * S) * HD;
  const __bf16* kbase = Kg + (((size_t)b * NKV + kvh) * S) * HD;
  const __bf16* vbase = Vg + (((size_t)b * NKV + kvh) * S) * HD;

  // Async-load the Q tile once (8 x 16B per thread).
#pragma unroll
  for (int it = 0; it < 8; ++it) {
    int slot = it * 256 + tid;
    int r = slot >> 4, c8 = (slot & 15) * 8;
    async_copy16(qbase + (size_t)(q0 + r) * HD + c8, Qs + r * FP + c8);
  }
  if (tid < 128) { mrow[tid] = -3.0e38f; lrow[tid] = 0.0f; }
  v8f o[8] = {};
  wait_async0();
  __syncthreads();

  for (int jt = 0; jt <= qt; ++jt) {
    int j0 = jt * 128;
    // K tile [key][hd] via async DMA; V tile transposed manually -> Vt[hd][key].
#pragma unroll
    for (int it = 0; it < 8; ++it) {
      int slot = it * 256 + tid;
      int r = slot >> 4, c8 = (slot & 15) * 8;
      async_copy16(kbase + (size_t)(j0 + r) * HD + c8, Ks + r * FP + c8);
    }
#pragma unroll
    for (int it = 0; it < 8; ++it) {
      int slot = it * 256 + tid;
      int r = slot >> 4, c8 = (slot & 15) * 8;
      const __bf16* src = vbase + (size_t)(j0 + r) * HD + c8;
#pragma unroll
      for (int j = 0; j < 8; ++j) Vt[(c8 + j) * FP + r] = src[j];
    }
    wait_async0();
    __syncthreads();

    // S = Q K^T : contraction over HD (4 WMMA k-steps of 32).
    v8f sa[8] = {};
#pragma unroll
    for (int ks = 0; ks < 4; ++ks) {
      int kb = ks * 32 + hf * 8;
      v16bf af = ld_frag(Qs + (16 * w + r16) * FP, kb);
      v16bf bfr[8];
#pragma unroll
      for (int nt = 0; nt < 8; ++nt)
        bfr[nt] = ld_frag(Ks + (16 * nt + r16) * FP, kb);
#pragma unroll
      for (int nt = 0; nt < 8; ++nt)
        sa[nt] = wmma_bf16(af, bfr[nt], sa[nt]);
    }
    bool diag = (jt == qt);
#pragma unroll
    for (int nt = 0; nt < 8; ++nt)
#pragma unroll
      for (int r = 0; r < 8; ++r) {
        int row = 16 * w + r + 8 * hf;
        int col = 16 * nt + r16;
        float v = sa[nt][r] * SCALE;
        if (diag && (j0 + col > q0 + row)) v = -3.0e38f;
        Sc[row * CP + col] = v;
      }
    __syncthreads();

    // Online softmax (one thread per query row).
    if (tid < 128) {
      float* sr = Sc + tid * CP;
      float mo = mrow[tid];
      float mt = mo;
      for (int c = 0; c < 128; ++c) mt = fmaxf(mt, sr[c]);
      float scale = __expf(mo - mt);
      float sum = 0.0f;
      __bf16* pr = Ps + tid * FP;
      for (int c = 0; c < 128; ++c) {
        float p = __expf(sr[c] - mt);
        pr[c] = (__bf16)p;
        sum += p;
      }
      mrow[tid] = mt;
      lrow[tid] = lrow[tid] * scale + sum;
      srow[tid] = scale;
    }
    __syncthreads();

    // O = O*scale + P.V : contraction over 128 keys (4 WMMA k-steps).
    float f[8];
#pragma unroll
    for (int r = 0; r < 8; ++r) f[r] = srow[16 * w + r + 8 * hf];
#pragma unroll
    for (int nt = 0; nt < 8; ++nt)
#pragma unroll
      for (int r = 0; r < 8; ++r) o[nt][r] *= f[r];
#pragma unroll
    for (int ks = 0; ks < 4; ++ks) {
      int kb = ks * 32 + hf * 8;
      v16bf af = ld_frag(Ps + (16 * w + r16) * FP, kb);
      v16bf bfr[8];
#pragma unroll
      for (int nt = 0; nt < 8; ++nt)
        bfr[nt] = ld_frag(Vt + (16 * nt + r16) * FP, kb);
#pragma unroll
      for (int nt = 0; nt < 8; ++nt)
        o[nt] = wmma_bf16(af, bfr[nt], o[nt]);
    }
    __syncthreads();
  }

  float fi[8];
#pragma unroll
  for (int r = 0; r < 8; ++r) fi[r] = 1.0f / lrow[16 * w + r + 8 * hf];
#pragma unroll
  for (int nt = 0; nt < 8; ++nt)
#pragma unroll
    for (int r = 0; r < 8; ++r) {
      int row = q0 + 16 * w + r + 8 * hf;
      int hd = 16 * nt + r16;
      Og[(((size_t)b * S + row) * NH + h) * HD + hd] = (__bf16)(o[nt][r] * fi[r]);
    }
}

// ---------------------------------------------------------------------------
// Kernel 3: output projection, attn[M,4096](bf16) x Wo[4096,HID] -> fp32 d_out,
// double-buffered LDS.
// ---------------------------------------------------------------------------
__global__ __launch_bounds__(256) void out_gemm_kernel(
    const __bf16* __restrict__ Ag, const float* __restrict__ Wg,
    float* __restrict__ outg) {
  extern __shared__ char smem[];
  __bf16* As0 = (__bf16*)smem;      // [2][128][AP]
  __bf16* Bs0 = As0 + 2 * TILE;     // [2][128 cols][AP]

  const int tid = threadIdx.x;
  const int w = tid >> 5, lane = tid & 31;
  const int hf = lane >> 4, r16 = lane & 15;
  const int m0 = blockIdx.x * 128, n0 = blockIdx.y * 128;
  const int K = NH * HD;  // 4096

  v8f acc[8] = {};

  auto load_tiles = [&](int k0, int buf) {
    __bf16* As = As0 + buf * TILE;
    __bf16* Bs = Bs0 + buf * TILE;
#pragma unroll
    for (int it = 0; it < 2; ++it) {
      int slot = it * 256 + tid;
      int r = slot >> 2, c8 = (slot & 3) * 8;
      const __bf16* src = Ag + (size_t)(m0 + r) * K + k0 + c8;
      __bf16* dst = As + r * AP + c8;
#pragma unroll
      for (int j = 0; j < 8; ++j) dst[j] = src[j];
    }
#pragma unroll
    for (int it = 0; it < 4; ++it) {
      int slot = it * 256 + tid;
      int kk = slot >> 5, c4 = (slot & 31) * 4;
      float4 v = *(const float4*)(Wg + (size_t)(k0 + kk) * HID + n0 + c4);
      Bs[(c4 + 0) * AP + kk] = (__bf16)v.x;
      Bs[(c4 + 1) * AP + kk] = (__bf16)v.y;
      Bs[(c4 + 2) * AP + kk] = (__bf16)v.z;
      Bs[(c4 + 3) * AP + kk] = (__bf16)v.w;
    }
    if (k0 + 32 < K)
      __builtin_prefetch(Wg + (size_t)(k0 + 32) * HID + n0, 0, 1);
  };

  auto compute = [&](int buf) {
    const __bf16* As = As0 + buf * TILE;
    const __bf16* Bs = Bs0 + buf * TILE;
    const int kb = hf * 8;
    v16bf af = ld_frag(As + (16 * w + r16) * AP, kb);
    v16bf bfr[8];
#pragma unroll
    for (int nt = 0; nt < 8; ++nt)
      bfr[nt] = ld_frag(Bs + (16 * nt + r16) * AP, kb);
#pragma unroll
    for (int nt = 0; nt < 8; ++nt)
      acc[nt] = wmma_bf16(af, bfr[nt], acc[nt]);
  };

  load_tiles(0, 0);
  int buf = 0;
  for (int k0 = 0; k0 < K; k0 += 32) {
    __syncthreads();
    if (k0 + 32 < K) load_tiles(k0 + 32, buf ^ 1);
    compute(buf);
    buf ^= 1;
  }

#pragma unroll
  for (int nt = 0; nt < 8; ++nt)
#pragma unroll
    for (int r = 0; r < 8; ++r) {
      int row = m0 + 16 * w + r + 8 * hf;
      outg[(size_t)row * HID + n0 + 16 * nt + r16] = acc[nt][r];
    }
}

extern "C" void kernel_launch(void* const* d_in, const int* in_sizes, int n_in,
                              void* d_out, int out_size, void* d_ws, size_t ws_size,
                              hipStream_t stream) {
  (void)in_sizes; (void)n_in; (void)out_size; (void)ws_size;
  const float* hs  = (const float*)d_in[0];
  const float* Wq  = (const float*)d_in[1];
  const float* Wk  = (const float*)d_in[2];
  const float* Wv  = (const float*)d_in[3];
  const float* Wo  = (const float*)d_in[4];
  const float* qnw = (const float*)d_in[5];
  const float* knw = (const float*)d_in[6];
  const int*   pos = (const int*)d_in[7];
  float* out = (float*)d_out;

  __bf16* qws = (__bf16*)d_ws;                      // [B,NH,S,HD]
  __bf16* kws = qws + (size_t)B * NH * S * HD;      // [B,NKV,S,HD]
  __bf16* vws = kws + (size_t)B * NKV * S * HD;     // [B,NKV,S,HD]
  __bf16* aws = vws + (size_t)B * NKV * S * HD;     // [B,S,NH,HD]

  const size_t smem_gemm      = (size_t)4 * TILE * sizeof(__bf16);  // 2x(A+B)
  const size_t smem_gemm_norm = smem_gemm + (size_t)128 * CP * sizeof(float);
  const size_t smem_attn = (size_t)3 * 128 * FP * sizeof(__bf16)   // Qs,Ks,Vt
                         + (size_t)128 * CP * sizeof(float)         // Sc
                         + (size_t)128 * FP * sizeof(__bf16)        // Ps
                         + (size_t)3 * 128 * sizeof(float);         // m,l,scale

  qkv_gemm_kernel<<<dim3(M / 128, NH), 256, smem_gemm_norm, stream>>>(
      hs, Wq, qnw, pos, qws, NH, 0);
  qkv_gemm_kernel<<<dim3(M / 128, NKV), 256, smem_gemm_norm, stream>>>(
      hs, Wk, knw, pos, kws, NKV, 1);
  qkv_gemm_kernel<<<dim3(M / 128, NKV), 256, smem_gemm, stream>>>(
      hs, Wv, nullptr, pos, vws, NKV, 2);
  attn_kernel<<<dim3(S / 128, NH, B), 256, smem_attn, stream>>>(qws, kws, vws, aws);
  out_gemm_kernel<<<dim3(M / 128, HID / 128), 256, smem_gemm, stream>>>(aws, Wo, out);
}
